// ProbAttention_858993459686
// MI455X (gfx1250) — matrix operand order repaired and checked by reading, not compiled, not run on hardware
//
#include <hip/hip_runtime.h>
#include <hip/hip_bf16.h>

// ---------------- problem constants ----------------
static constexpr int B  = 4;
static constexpr int L  = 4096;
static constexpr int H  = 8;
static constexpr int D  = 64;
static constexpr int SK = 45;      // sample_k = n_top = min(5*ceil(ln 4096), 4096)
static constexpr int NT = 45;      // number of top queries kept per (b,h)
static constexpr int NTP = 48;     // padded to 3 WMMA tiles of 16
static constexpr int NCHUNK = 16;  // cumsum chunks
static constexpr int CHUNK  = L / NCHUNK;  // 256
static constexpr int NWAVE  = 4;   // waves per attention workgroup (key-split)
static constexpr int KEYS_PER_WAVE = L / NWAVE;   // 1024

typedef __attribute__((ext_vector_type(16))) _Float16 v16h;
typedef __attribute__((ext_vector_type(8)))  float    v8f;

union H16 { v16h v; unsigned int u[8]; _Float16 h[16]; };
union P2  { unsigned int u; _Float16 h[2]; };

#define NEG_INF (-__builtin_inff())

// =====================================================================
// Kernel 1: sparsity metric  M[b,h,l] = max_j(q.k_sample_j) - sum_j/L
// =====================================================================
__global__ __launch_bounds__(256)
void mscore_kernel(const float* __restrict__ q, const float* __restrict__ k,
                   const int* __restrict__ idx, float* __restrict__ Mout) {
    int tid = blockIdx.x * 256 + threadIdx.x;          // tid = (b*H + h)*L + l
    if (tid >= B * H * L) return;
    int l  = tid % L;
    int bh = tid / L;
    int h  = bh % H;
    int b  = bh / H;

    const float* qrow = q + ((size_t)(b * L + l) * H + h) * D;
    float4 qv[16];
#pragma unroll
    for (int i = 0; i < 16; ++i) qv[i] = *(const float4*)(qrow + 4 * i);

    float mx = NEG_INF, sm = 0.0f;
    for (int j = 0; j < SK; ++j) {
        int kid = idx[l * SK + j];
        const float* krow = k + ((size_t)(b * L + kid) * H + h) * D;
        float acc = 0.0f;
#pragma unroll
        for (int i = 0; i < 16; ++i) {
            float4 kv = *(const float4*)(krow + 4 * i);
            acc += qv[i].x * kv.x + qv[i].y * kv.y + qv[i].z * kv.z + qv[i].w * kv.w;
        }
        mx = fmaxf(mx, acc);
        sm += acc;
    }
    Mout[tid] = mx - sm * (1.0f / (float)L);
}

// =====================================================================
// Kernel 2: per (b,h) top-45 query positions (iterative argmax in LDS)
// =====================================================================
__global__ __launch_bounds__(256)
void topk_kernel(const float* __restrict__ M, int* __restrict__ top) {
    __shared__ float vals[L];
    __shared__ float smax[256];
    __shared__ int   sidx[256];
    int bh = blockIdx.x, t = threadIdx.x;
    const float* row = M + (size_t)bh * L;
    for (int i = t; i < L; i += 256) vals[i] = row[i];
    __syncthreads();

    for (int it = 0; it < NT; ++it) {
        float bm = NEG_INF; int bi = 0;
        for (int i = t; i < L; i += 256) {
            float v = vals[i];
            if (v > bm) { bm = v; bi = i; }
        }
        smax[t] = bm; sidx[t] = bi;
        __syncthreads();
        for (int s = 128; s > 0; s >>= 1) {
            if (t < s && smax[t + s] > smax[t]) { smax[t] = smax[t + s]; sidx[t] = sidx[t + s]; }
            __syncthreads();
        }
        if (t == 0) { top[bh * NTP + it] = sidx[0]; vals[sidx[0]] = NEG_INF; }
        __syncthreads();
    }
    if (t < NTP - NT) top[bh * NTP + NT + t] = 0;   // padded rows -> qpos 0 (never stored)
}

// =====================================================================
// Kernel 3a/3b: out[b,h,l,d] = cumsum_l V[b,l,h,d]  (chunked 2-pass scan)
// =====================================================================
__global__ __launch_bounds__(64)
void cumsum_a(const float* __restrict__ v, float* __restrict__ out, float* __restrict__ csum) {
    int bid = blockIdx.x;             // (b*H+h)*NCHUNK + chunk
    int bh = bid / NCHUNK, chunk = bid % NCHUNK;
    int b = bh / H, h = bh % H;
    int d = threadIdx.x;
    float acc = 0.0f;
    for (int i = 0; i < CHUNK; ++i) {
        int l = chunk * CHUNK + i;
        acc += v[((size_t)(b * L + l) * H + h) * D + d];
        out[((size_t)bh * L + l) * D + d] = acc;
    }
    csum[(size_t)bid * D + d] = acc;
}

__global__ __launch_bounds__(64)
void cumsum_b(float* __restrict__ out, const float* __restrict__ csum) {
    int bid = blockIdx.x;
    int bh = bid / NCHUNK, chunk = bid % NCHUNK;
    int d = threadIdx.x;
    if (chunk == 0) return;
    float pre = 0.0f;
    for (int c = 0; c < chunk; ++c) pre += csum[(size_t)(bh * NCHUNK + c) * D + d];
    for (int i = 0; i < CHUNK; ++i) {
        int l = chunk * CHUNK + i;
        out[((size_t)bh * L + l) * D + d] += pre;
    }
}

// =====================================================================
// Kernel 4: flash-style causal attention for the top queries via WMMA.
// 4 waves per (b, h, 16-query tile); each wave owns a 1024-key range and
// private LDS tiles, producing flash partials (m, l, O) that are merged
// through LDS at the end.  Per wave, keys stream in 32-wide blocks:
//   scores: D = Aq(16x32 f16) x Kt^T(32x16 f16) + C   (2 N-tiles x 2 K-chunks)
//   PV:     O = P(16x32 f16) x V(32x64 f16) + O       (4 N-tiles)
// =====================================================================
__global__ __launch_bounds__(32 * NWAVE)
void attn_kernel(const float* __restrict__ q, const float* __restrict__ k,
                 const float* __restrict__ v, const int* __restrict__ top,
                 float* __restrict__ out) {
    // Per-wave LDS tiles (dword typed, half-addressed via cast). Rows padded
    // to odd dword strides to avoid bank conflicts.
    __shared__ unsigned int kt[NWAVE][32 * 33];  // K tile: [key][dim] f16
    __shared__ unsigned int vt[NWAVE][64 * 17];  // V tile transposed: [dim][key] f16
    __shared__ unsigned int pt[NWAVE][16 * 17];  // P tile: [qrow][key] f16
    __shared__ float osh[NWAVE][16 * 64];        // per-wave partial O
    __shared__ float msh[NWAVE][16];             // per-wave row max
    __shared__ float lsh[NWAVE][16];             // per-wave row sum
    __shared__ int qpos_s[16];

    const int bid   = blockIdx.x;              // (b*H+h)*3 + mtile
    const int bh    = bid / 3;
    const int mtile = bid % 3;
    const int b     = bh / H;
    const int h     = bh % H;
    const int wave  = threadIdx.x >> 5;
    const int lane  = threadIdx.x & 31;
    const int mrow  = lane & 15;               // M index for A operands / N index for C
    const int half  = lane >> 4;               // lane group (0: lanes 0-15, 1: 16-31)

    _Float16* vth = (_Float16*)vt[wave];
    _Float16* pth = (_Float16*)pt[wave];

    if (threadIdx.x < 16) qpos_s[threadIdx.x] = top[bh * NTP + mtile * 16 + threadIdx.x];
    __syncthreads();

    // per-lane row metadata (rows r+8*half for r=0..7, matching C/D layout)
    int   qp8[8];
    float mrun[8], lrun[8];
#pragma unroll
    for (int r = 0; r < 8; ++r) {
        qp8[r]  = qpos_s[r + 8 * half];
        mrun[r] = NEG_INF;
        lrun[r] = 0.0f;
    }

    // Load Q tile into A-operand registers (two K-chunks of 32 dims)
    H16 aq[2];
    {
        const float* qrow = q + ((size_t)(b * L + qpos_s[mrow]) * H + h) * D;
#pragma unroll
        for (int ch = 0; ch < 2; ++ch)
#pragma unroll
            for (int vv = 0; vv < 8; ++vv) {
                int kb = ch * 32 + (vv >> 2) * 16 + half * 8 + (vv & 3) * 2;
                P2 pp;
                pp.h[0] = (_Float16)qrow[kb];
                pp.h[1] = (_Float16)qrow[kb + 1];
                aq[ch].u[vv] = pp.u;
            }
    }

    const float SCALE = 0.125f;  // 1/sqrt(64)
    v8f o[4];
#pragma unroll
    for (int j = 0; j < 4; ++j) o[j] = (v8f){0.f,0.f,0.f,0.f,0.f,0.f,0.f,0.f};

    for (int blk = 0; blk < KEYS_PER_WAVE / 32; ++blk) {
        const int kbase = wave * KEYS_PER_WAVE + blk * 32;
        // ---- stage K (natural) and V (transposed) tiles as f16 in LDS ----
        {
            const int kidx = kbase + lane;
            const float* krow = k + ((size_t)(b * L + kidx) * H + h) * D;
            const float* vrow = v + ((size_t)(b * L + kidx) * H + h) * D;
#pragma unroll
            for (int d4 = 0; d4 < 16; ++d4) {
                float4 kq = *(const float4*)(krow + d4 * 4);
                P2 p0, p1;
                p0.h[0] = (_Float16)kq.x; p0.h[1] = (_Float16)kq.y;
                p1.h[0] = (_Float16)kq.z; p1.h[1] = (_Float16)kq.w;
                kt[wave][lane * 33 + d4 * 2]     = p0.u;
                kt[wave][lane * 33 + d4 * 2 + 1] = p1.u;
                float4 vq = *(const float4*)(vrow + d4 * 4);
                vth[(d4 * 4 + 0) * 34 + lane] = (_Float16)vq.x;
                vth[(d4 * 4 + 1) * 34 + lane] = (_Float16)vq.y;
                vth[(d4 * 4 + 2) * 34 + lane] = (_Float16)vq.z;
                vth[(d4 * 4 + 3) * 34 + lane] = (_Float16)vq.w;
            }
        }

        // ---- scores S(16x32) = Q x K^T via 4 WMMAs ----
        float s[2][8];
#pragma unroll
        for (int n = 0; n < 2; ++n) {
            v8f c = (v8f){0.f,0.f,0.f,0.f,0.f,0.f,0.f,0.f};
#pragma unroll
            for (int ch = 0; ch < 2; ++ch) {
                H16 bk;
#pragma unroll
                for (int vv = 0; vv < 8; ++vv)
                    bk.u[vv] = kt[wave][(n * 16 + mrow) * 33 + ch * 16 + (vv >> 2) * 8 + half * 4 + (vv & 3)];
                c = __builtin_amdgcn_wmma_f32_16x16x32_f16(
                        false, aq[ch].v, false, bk.v, (short)0, c, false, false);
            }
            const int kpos = kbase + n * 16 + mrow;   // N = lane%16
#pragma unroll
            for (int r = 0; r < 8; ++r)
                s[n][r] = (kpos > qp8[r]) ? NEG_INF : c[r] * SCALE;  // causal mask
        }

        // ---- online softmax update ----
        float alpha[8];
#pragma unroll
        for (int r = 0; r < 8; ++r) {
            float t = fmaxf(s[0][r], s[1][r]);
            t = fmaxf(t, __shfl_xor(t, 1));
            t = fmaxf(t, __shfl_xor(t, 2));
            t = fmaxf(t, __shfl_xor(t, 4));
            t = fmaxf(t, __shfl_xor(t, 8));
            float mnew = fmaxf(mrun[r], t);
            float a = (mnew == NEG_INF) ? 1.0f : __expf(mrun[r] - mnew);
            mrun[r] = mnew;
            float p0 = __expf(s[0][r] - mnew);   // masked (-inf) -> 0
            float p1 = __expf(s[1][r] - mnew);
            float ps = p0 + p1;
            ps += __shfl_xor(ps, 1);
            ps += __shfl_xor(ps, 2);
            ps += __shfl_xor(ps, 4);
            ps += __shfl_xor(ps, 8);
            lrun[r] = lrun[r] * a + ps;
            alpha[r] = a;
            // C-layout -> LDS (row M = r+8*half, col = n*16 + lane%16)
            pth[(r + 8 * half) * 34 + mrow]      = (_Float16)p0;
            pth[(r + 8 * half) * 34 + 16 + mrow] = (_Float16)p1;
        }
#pragma unroll
        for (int j = 0; j < 4; ++j)
#pragma unroll
            for (int r = 0; r < 8; ++r) o[j][r] *= alpha[r];

        // ---- read P back in A-layout ----
        H16 ap;
#pragma unroll
        for (int vv = 0; vv < 8; ++vv)
            ap.u[vv] = pt[wave][mrow * 17 + (vv >> 2) * 8 + half * 4 + (vv & 3)];

        // ---- O += P x V : 4 WMMAs over 16-dim N-tiles ----
#pragma unroll
        for (int j = 0; j < 4; ++j) {
            H16 bv;
#pragma unroll
            for (int vv = 0; vv < 8; ++vv)
                bv.u[vv] = vt[wave][(j * 16 + mrow) * 17 + (vv >> 2) * 8 + half * 4 + (vv & 3)];
            o[j] = __builtin_amdgcn_wmma_f32_16x16x32_f16(
                       false, ap.v, false, bv.v, (short)0, o[j], false, false);
        }
    }

    // ---- publish per-wave flash partials (m, l, O) to LDS ----
    if (mrow == 0) {
#pragma unroll
        for (int r = 0; r < 8; ++r) {
            msh[wave][r + 8 * half] = mrun[r];
            lsh[wave][r + 8 * half] = lrun[r];
        }
    }
#pragma unroll
    for (int r = 0; r < 8; ++r)
#pragma unroll
        for (int j = 0; j < 4; ++j)
            osh[wave][(r + 8 * half) * 64 + j * 16 + mrow] = o[j][r];
    __syncthreads();

    // ---- merge partials across waves and scatter the top-query rows ----
    for (int e = threadIdx.x; e < 16 * 64; e += 32 * NWAVE) {
        int row = e >> 6;
        int dim = e & 63;
        int u = mtile * 16 + row;
        if (u >= NT) continue;                  // padded rows
        float mt = msh[0][row];
#pragma unroll
        for (int w = 1; w < NWAVE; ++w) mt = fmaxf(mt, msh[w][row]);
        float lt = 0.0f, val = 0.0f;
#pragma unroll
        for (int w = 0; w < NWAVE; ++w) {
            float sc = __expf(msh[w][row] - mt);   // wave 0 has key 0 -> mt finite
            lt  += sc * lsh[w][row];
            val += sc * osh[w][row * 64 + dim];
        }
        out[((size_t)bh * L + qpos_s[row]) * D + dim] = val / lt;
    }
}

// =====================================================================
extern "C" void kernel_launch(void* const* d_in, const int* in_sizes, int n_in,
                              void* d_out, int out_size, void* d_ws, size_t ws_size,
                              hipStream_t stream) {
    const float* q   = (const float*)d_in[0];
    const float* k   = (const float*)d_in[1];
    const float* v   = (const float*)d_in[2];
    const int*   idx = (const int*)d_in[3];
    float* out = (float*)d_out;

    char* ws = (char*)d_ws;
    float* wsM   = (float*)ws;                                   // B*H*L floats
    int*   wsTop = (int*)(ws + (size_t)B * H * L * 4);           // B*H*48 ints
    float* wsC   = (float*)(ws + (size_t)B * H * L * 4 +
                                 (size_t)B * H * NTP * 4);       // B*H*16*64 floats

    mscore_kernel<<<(B * H * L) / 256, 256, 0, stream>>>(q, k, idx, wsM);
    topk_kernel<<<B * H, 256, 0, stream>>>(wsM, wsTop);
    cumsum_a<<<B * H * NCHUNK, 64, 0, stream>>>(v, out, wsC);
    cumsum_b<<<B * H * NCHUNK, 64, 0, stream>>>(out, wsC);
    attn_kernel<<<B * H * 3, 32 * NWAVE, 0, stream>>>(q, k, v, wsTop, out);
}
